// CovariantEvolutionBlock_54494545051808
// MI455X (gfx1250) — matrix operand order repaired and checked by reading, not compiled, not run on hardware
//
#include <hip/hip_runtime.h>
#include <hip/hip_fp16.h>

typedef _Float16 h16;
typedef __attribute__((ext_vector_type(16))) _Float16 v16h;
typedef __attribute__((ext_vector_type(8)))  float    v8f;
typedef __attribute__((ext_vector_type(4)))  int      i4;

#define D_    1024
#define NTOK  4096
#define LSEQ  2048
#define NH    16
#define HD    64

// ---------------------------------------------------------------------------
// CDNA5 async global->LDS staging (ASYNCcnt) with compile-safe fallback
// ---------------------------------------------------------------------------
#if defined(__gfx1250__)
#if __has_builtin(__builtin_amdgcn_global_load_async_to_lds_b128)
#define HAVE_ASYNC_LDS 1
#else
#define HAVE_ASYNC_LDS 0
#endif
#else
#define HAVE_ASYNC_LDS 0
#endif

#if HAVE_ASYNC_LDS
typedef __attribute__((address_space(1))) i4* gp_i4;  // global 16B vector ptr
typedef __attribute__((address_space(3))) i4* lp_i4;  // LDS    16B vector ptr
#endif

__device__ __forceinline__ void ld_async_b128(h16* lds, const h16* g) {
#if HAVE_ASYNC_LDS
  __builtin_amdgcn_global_load_async_to_lds_b128(
      (gp_i4)(i4*)(void*)const_cast<h16*>(g),
      (lp_i4)(i4*)(void*)lds, 0, 0);
#else
  *(uint4*)lds = *(const uint4*)g;
#endif
}

__device__ __forceinline__ void wait_async() {
#if HAVE_ASYNC_LDS
#if __has_builtin(__builtin_amdgcn_s_wait_asynccnt)
  __builtin_amdgcn_s_wait_asynccnt(0);
#else
  asm volatile("s_wait_asynccnt 0" ::: "memory");
#endif
#endif
}

__device__ __forceinline__ float fast_rcp(float x) {
#if defined(__gfx1250__) && __has_builtin(__builtin_amdgcn_rcpf)
  return __builtin_amdgcn_rcpf(x);
#else
  return 1.f / x;
#endif
}

// ---------------------------------------------------------------------------
// fp32 -> fp16 convert
// ---------------------------------------------------------------------------
__global__ void cvt_kernel(const float* __restrict__ x, h16* __restrict__ o, int n) {
  int i = blockIdx.x * 256 + threadIdx.x;
  if (i < n) o[i] = (h16)x[i];
}

// ---------------------------------------------------------------------------
// RMSNorm with up to two f16 destinations (second used to fuse the cn copy)
// one row (D=1024) per 256-thread block
// ---------------------------------------------------------------------------
__global__ __launch_bounds__(256)
void rms_kernel(const float* __restrict__ x, const float* __restrict__ w,
                h16* __restrict__ out, int ldo, int colOff,
                h16* __restrict__ out2, int ldo2, int colOff2) {
  __shared__ float red[8];
  const int row = blockIdx.x;
  const int tid = threadIdx.x;
  const float* xr = x + (size_t)row * D_;
  float xv[4];
  float ss = 0.f;
#pragma unroll
  for (int i = 0; i < 4; ++i) { xv[i] = xr[tid + i * 256]; ss += xv[i] * xv[i]; }
#pragma unroll
  for (int m = 16; m >= 1; m >>= 1) ss += __shfl_xor(ss, m);
  if ((tid & 31) == 0) red[tid >> 5] = ss;
  __syncthreads();
  float tot = 0.f;
#pragma unroll
  for (int i = 0; i < 8; ++i) tot += red[i];
  const float sc = rsqrtf(tot * (1.0f / D_) + 1e-6f);
#pragma unroll
  for (int i = 0; i < 4; ++i) {
    const int c = tid + i * 256;
    const h16 v = (h16)(xv[i] * sc * w[c]);
    out[(size_t)row * ldo + colOff + c] = v;
    if (out2) out2[(size_t)row * ldo2 + colOff2 + c] = v;
  }
}

// ---------------------------------------------------------------------------
// Generic WMMA GEMM: Y[row, col] = act( X[row,:K] . W[col,:K] + bias[col] ) (+res)
// X: f16, row stride lda (halves). W: f16 [M, K] row-major (K contiguous).
// Block tile 64 rows x 128 cols; 8 waves, each 2x2 tiles of 16x16.
// LDS staged in K-chunks of 64 via async global->LDS; 8 WMMAs per barrier pair.
// ACT: 0=none 1=silu 2=tanh
// ---------------------------------------------------------------------------
template <int ACT>
__global__ __launch_bounds__(256)
void gemm16(const h16* __restrict__ X, int lda,
            const h16* __restrict__ W, int K,
            const float* __restrict__ bias,
            float* __restrict__ out32, int ldo32,
            h16* __restrict__ out16, int ldo16,
            const float* __restrict__ residual) {
  __shared__ h16 Xs[64][72];
  __shared__ h16 Ws[128][72];
  const int tid  = threadIdx.x;
  const int lane = tid & 31;
  const int wave = tid >> 5;
  const int wr = wave >> 2;   // 0..1 : wave row
  const int wc = wave & 3;    // 0..3 : wave col
  const int rowBase = blockIdx.y * 64;
  const int colBase = blockIdx.x * 128;

  v8f acc[2][2] = {};

  const int xr  = tid >> 2, xc  = (tid & 3) * 16;  // X tile: 64x64 halves
  const int wrw = tid >> 1, wcg = (tid & 1) * 32;  // W tile: 128x64 halves

  for (int k0 = 0; k0 < K; k0 += 64) {
    const h16* xp = X + (size_t)(rowBase + xr) * lda + k0 + xc;
    ld_async_b128(&Xs[xr][xc],     xp);
    ld_async_b128(&Xs[xr][xc + 8], xp + 8);
    const h16* wp = W + (size_t)(colBase + wrw) * K + k0 + wcg;
    ld_async_b128(&Ws[wrw][wcg],      wp);
    ld_async_b128(&Ws[wrw][wcg + 8],  wp + 8);
    ld_async_b128(&Ws[wrw][wcg + 16], wp + 16);
    ld_async_b128(&Ws[wrw][wcg + 24], wp + 24);
    wait_async();
    __syncthreads();

    const int m = lane & 15;
#pragma unroll
    for (int ks = 0; ks < 2; ++ks) {
      union { uint4 u[2]; v16h v; } fa[2], fb[2];
      const int ka  = ks * 32 + ((lane < 16) ? 0 : 8);    // A-frag K chunk base
      const int kbk = ks * 32 + ((lane < 16) ? 0 : 8);    // B-frag K chunk base
#pragma unroll
      for (int rt = 0; rt < 2; ++rt) {
        const h16* p = &Xs[wr * 32 + rt * 16 + m][ka];
        fa[rt].u[0] = *(const uint4*)p;
        fa[rt].u[1] = *(const uint4*)(p + 16);
      }
#pragma unroll
      for (int ct = 0; ct < 2; ++ct) {
        const h16* p = &Ws[wc * 32 + ct * 16 + m][kbk];
        fb[ct].u[0] = *(const uint4*)p;
        fb[ct].u[1] = *(const uint4*)(p + 16);
      }
#pragma unroll
      for (int rt = 0; rt < 2; ++rt)
#pragma unroll
        for (int ct = 0; ct < 2; ++ct)
          acc[rt][ct] = __builtin_amdgcn_wmma_f32_16x16x32_f16(
              false, fa[rt].v, false, fb[ct].v, (short)0, acc[rt][ct], false, false);
    }
    __syncthreads();
  }

  // epilogue: D layout -> rows g / g+8 across lane halves, cols = lane%16
  const int m   = lane & 15;
  const int rb2 = (lane < 16) ? 0 : 8;
#pragma unroll
  for (int rt = 0; rt < 2; ++rt)
#pragma unroll
    for (int ct = 0; ct < 2; ++ct) {
      const int col = colBase + wc * 32 + ct * 16 + m;
      const float bv = bias ? bias[col] : 0.f;
#pragma unroll
      for (int g = 0; g < 8; ++g) {
        const int row = rowBase + wr * 32 + rt * 16 + g + rb2;
        float v = acc[rt][ct][g] + bv;
        if (ACT == 1) v = v * fast_rcp(1.f + __expf(-v));
        else if (ACT == 2) v = tanhf(v);
        if (residual) v += residual[(size_t)row * ldo32 + col];
        if (out32) out32[(size_t)row * ldo32 + col] = v;
        if (out16) out16[(size_t)row * ldo16 + col] = (h16)v;
      }
    }
}

// ---------------------------------------------------------------------------
// V transpose: [N, D] -> [B, H, 64, L]
// ---------------------------------------------------------------------------
__global__ void vtrans_kernel(const h16* __restrict__ V, h16* __restrict__ Vt) {
  const size_t i = (size_t)blockIdx.x * 256 + threadIdx.x;  // over N*D
  const size_t row = i >> 10, col = i & 1023;
  const size_t b = row >> 11, l = row & 2047;
  Vt[((b * NH + (col >> 6)) * HD + (col & 63)) * (size_t)LSEQ + l] = V[i];
}

// ---------------------------------------------------------------------------
// Streaming sigmoid attention. One wave = one (b, h, 16-query tile).
// attn[l, h*64+d] = sum_m sigmoid(s)*V / max(1, sum_m sigmoid(s))
// ---------------------------------------------------------------------------
__global__ __launch_bounds__(256)
void attn_kernel(const h16* __restrict__ Q, const h16* __restrict__ Kc,
                 const h16* __restrict__ Vt, h16* __restrict__ attn,
                 const float* __restrict__ temp_ptr) {
  __shared__ h16 P[8][16][40];
  const int tid = threadIdx.x, lane = tid & 31, wave = tid >> 5;
  const int wid = blockIdx.x * 8 + wave;
  const int qt = wid & 127;
  const int h  = (wid >> 7) & 15;
  const int b  = wid >> 11;
  const int qRow = b * LSEQ + qt * 16;
  const float scale = 0.125f * temp_ptr[0];  // HD^-0.5 * temp

  union F { uint4 u[2]; v16h v; };
  const int m = lane & 15;

  F qa[2];
#pragma unroll
  for (int ds = 0; ds < 2; ++ds) {
    const h16* p = Q + (size_t)(qRow + m) * D_ + h * 64 + ds * 32 + ((lane < 16) ? 0 : 8);
    qa[ds].u[0] = *(const uint4*)p;
    qa[ds].u[1] = *(const uint4*)(p + 16);
  }

  v8f o[4] = {};
  float rs[8] = {};

  for (int kt = 0; kt < LSEQ / 32; ++kt) {
    const int keyBase = kt * 32;
    v8f s[2] = {};
#pragma unroll
    for (int ct = 0; ct < 2; ++ct)
#pragma unroll
      for (int ds = 0; ds < 2; ++ds) {
        F kb;
        const h16* p = Kc + (size_t)(b * LSEQ + keyBase + ct * 16 + m) * D_ +
                       h * 64 + ds * 32 + ((lane < 16) ? 0 : 8);
        kb.u[0] = *(const uint4*)p;
        kb.u[1] = *(const uint4*)(p + 16);
        s[ct] = __builtin_amdgcn_wmma_f32_16x16x32_f16(
            false, qa[ds].v, false, kb.v, (short)0, s[ct], false, false);
      }

    const int prow = (lane < 16) ? 0 : 8;
#pragma unroll
    for (int ct = 0; ct < 2; ++ct)
#pragma unroll
      for (int g = 0; g < 8; ++g) {
        const float sv = fast_rcp(1.f + __expf(-(s[ct][g] * scale)));
        rs[g] += sv;
        P[wave][g + prow][ct * 16 + m] = (h16)sv;
      }
    __builtin_amdgcn_wave_barrier();
    asm volatile("" ::: "memory");

    F pa;
    {
      const int ka = (lane < 16) ? 0 : 8;
      const h16* p = &P[wave][m][ka];
      pa.u[0] = *(const uint4*)p;
      pa.u[1] = *(const uint4*)(p + 16);
    }
#pragma unroll
    for (int dg = 0; dg < 4; ++dg) {
      F vb;
      const h16* p = Vt + ((size_t)(b * NH + h) * HD + dg * 16 + m) * (size_t)LSEQ +
                     keyBase + ((lane < 16) ? 0 : 8);
      vb.u[0] = *(const uint4*)p;
      vb.u[1] = *(const uint4*)(p + 16);
      o[dg] = __builtin_amdgcn_wmma_f32_16x16x32_f16(
          false, pa.v, false, vb.v, (short)0, o[dg], false, false);
    }
    __builtin_amdgcn_wave_barrier();
    asm volatile("" ::: "memory");
  }

  // full row sums: reduce across the 16-lane half (D layout rows g / g+8)
#pragma unroll
  for (int g = 0; g < 8; ++g) {
    float v = rs[g];
    v += __shfl_xor(v, 1);
    v += __shfl_xor(v, 2);
    v += __shfl_xor(v, 4);
    v += __shfl_xor(v, 8);
    rs[g] = fast_rcp(fmaxf(v, 1.0f));
  }
  const int prow = (lane < 16) ? 0 : 8;
#pragma unroll
  for (int dg = 0; dg < 4; ++dg)
#pragma unroll
    for (int g = 0; g < 8; ++g)
      attn[(size_t)(qRow + g + prow) * D_ + h * 64 + dg * 16 + m] =
          (h16)(o[dg][g] * rs[g]);
}

// ---------------------------------------------------------------------------
// dz = dt*(dzl+corr); z1 = z + dz + ctx; cu_in = [conn | z1 | dz] f16; out3 = z
// ---------------------------------------------------------------------------
__global__ void fuse1_kernel(const float* __restrict__ z, const float* __restrict__ conn,
                             const float* __restrict__ dzl, const float* __restrict__ corr,
                             const float* __restrict__ ctx, const float* __restrict__ dt_ptr,
                             float* __restrict__ z1, h16* __restrict__ cuin,
                             float* __restrict__ out3) {
  const size_t i = (size_t)blockIdx.x * 256 + threadIdx.x;  // over N*D
  const float dt = dt_ptr[0];
  const float dzv = dt * (dzl[i] + corr[i]);
  const float z1v = z[i] + dzv + ctx[i];
  z1[i] = z1v;
  const size_t row = i >> 10, col = i & 1023;
  h16* cr = cuin + row * 3072;
  cr[col]        = (h16)conn[i];
  cr[1024 + col] = (h16)z1v;
  cr[2048 + col] = (h16)dzv;
  out3[i] = z[i];
}

// ---------------------------------------------------------------------------
extern "C" void kernel_launch(void* const* d_in, const int* in_sizes, int n_in,
                              void* d_out, int out_size, void* d_ws, size_t ws_size,
                              hipStream_t stream) {
  (void)in_sizes; (void)n_in; (void)out_size; (void)ws_size;
  const float* z     = (const float*)d_in[0];
  const float* conn  = (const float*)d_in[1];
  const float* w_z   = (const float*)d_in[2];
  const float* w_c   = (const float*)d_in[3];
  const float* w_mlp = (const float*)d_in[4];
  const float* f_w1  = (const float*)d_in[5];
  const float* f_b1  = (const float*)d_in[6];
  const float* f_w2  = (const float*)d_in[7];
  const float* f_b2  = (const float*)d_in[8];
  const float* g_w1  = (const float*)d_in[9];
  const float* g_b1  = (const float*)d_in[10];
  const float* g_w2  = (const float*)d_in[11];
  const float* g_b2  = (const float*)d_in[12];
  const float* dt    = (const float*)d_in[13];
  const float* q_w   = (const float*)d_in[14];
  const float* k_w   = (const float*)d_in[15];
  const float* v_w   = (const float*)d_in[16];
  const float* o_w   = (const float*)d_in[17];
  const float* temp  = (const float*)d_in[18];
  const float* cu_w1 = (const float*)d_in[19];
  const float* cu_b1 = (const float*)d_in[20];
  const float* cu_w2 = (const float*)d_in[21];
  const float* cu_b2 = (const float*)d_in[22];
  const float* m_w1  = (const float*)d_in[23];
  const float* m_b1  = (const float*)d_in[24];
  const float* m_w2  = (const float*)d_in[25];
  const float* m_b2  = (const float*)d_in[26];

  float* out_z2 = (float*)d_out;
  float* out_cn = out_z2 + (size_t)NTOK * D_;
  float* out_zb = out_cn + (size_t)NTOK * D_;

  char* p = (char*)d_ws;
  auto a16 = [&](size_t n) { h16* r = (h16*)p; p += n * sizeof(h16); return r; };
  auto a32 = [&](size_t n) { float* r = (float*)p; p += n * sizeof(float); return r; };
  const size_t ND = (size_t)1024 * 1024;

  h16 *wf1 = a16(2 * ND), *wf2 = a16(2 * ND), *wg1 = a16(2 * ND), *wg2 = a16(ND),
      *wq  = a16(2 * ND), *wk  = a16(2 * ND), *wv  = a16(ND),     *wo  = a16(ND),
      *wc1 = a16(6 * ND), *wc2 = a16(2 * ND), *wm1 = a16(4 * ND), *wm2 = a16(4 * ND);
  h16 *zc   = a16((size_t)NTOK * 2048), *hbig = a16((size_t)NTOK * 4096),
      *gin  = a16((size_t)NTOK * 2048), *cuin = a16((size_t)NTOK * 3072),
      *qb   = a16((size_t)NTOK * 1024), *kb   = a16((size_t)NTOK * 1024),
      *vb   = a16((size_t)NTOK * 1024), *vt   = a16((size_t)NTOK * 1024),
      *atb  = a16((size_t)NTOK * 1024), *z1n  = a16((size_t)NTOK * 1024);
  float *dzl  = a32((size_t)NTOK * 1024), *corr = a32((size_t)NTOK * 1024),
        *ctx  = a32((size_t)NTOK * 1024), *z1   = a32((size_t)NTOK * 1024);

  auto cvt = [&](const float* s, h16* d, size_t n) {
    cvt_kernel<<<dim3((unsigned)((n + 255) / 256)), 256, 0, stream>>>(s, d, (int)n);
  };
  cvt(f_w1, wf1, 2 * ND);  cvt(f_w2, wf2, 2 * ND);
  cvt(g_w1, wg1, 2 * ND);  cvt(g_w2, wg2, ND);
  cvt(q_w,  wq,  2 * ND);  cvt(k_w,  wk,  2 * ND);
  cvt(v_w,  wv,  ND);      cvt(o_w,  wo,  ND);
  cvt(cu_w1, wc1, 6 * ND); cvt(cu_w2, wc2, 2 * ND);
  cvt(m_w1, wm1, 4 * ND);  cvt(m_w2, wm2, 4 * ND);

  auto gemm = [&](int act, const h16* X, int lda, const h16* W, int K,
                  const float* bias, int M, float* o32, int ldo32,
                  h16* o16, int ldo16, const float* res) {
    dim3 g(M / 128, NTOK / 64);
    if (act == 0)      gemm16<0><<<g, 256, 0, stream>>>(X, lda, W, K, bias, o32, ldo32, o16, ldo16, res);
    else if (act == 1) gemm16<1><<<g, 256, 0, stream>>>(X, lda, W, K, bias, o32, ldo32, o16, ldo16, res);
    else               gemm16<2><<<g, 256, 0, stream>>>(X, lda, W, K, bias, o32, ldo32, o16, ldo16, res);
  };

  // zn, cn -> zc (f16, stride 2048); cn additionally into g_in[:, 0:D]
  rms_kernel<<<NTOK, 256, 0, stream>>>(z,    w_z, zc, 2048, 0,    nullptr, 0, 0);
  rms_kernel<<<NTOK, 256, 0, stream>>>(conn, w_c, zc, 2048, 1024, gin, 2048, 0);

  // dz_local path
  gemm(1, zc,   2048, wf1, 1024, f_b1, 2048, nullptr, 0,    hbig,       2048, nullptr);
  gemm(0, hbig, 2048, wf2, 2048, f_b2, 1024, dzl,     1024, gin + 1024, 2048, nullptr);
  gemm(2, gin,  2048, wg1, 2048, g_b1, 1024, nullptr, 0,    hbig,       1024, nullptr);
  gemm(0, hbig, 1024, wg2, 1024, g_b2, 1024, corr,    1024, nullptr,    0,    nullptr);

  // attention path
  gemm(0, zc, 2048, wq, 2048, nullptr, 1024, nullptr, 0, qb, 1024, nullptr);
  gemm(0, zc, 2048, wk, 2048, nullptr, 1024, nullptr, 0, kb, 1024, nullptr);
  gemm(0, zc, 2048, wv, 1024, nullptr, 1024, nullptr, 0, vb, 1024, nullptr);  // zn only (K=1024)
  vtrans_kernel<<<16384, 256, 0, stream>>>(vb, vt);
  attn_kernel<<<512, 256, 0, stream>>>(qb, kb, vt, atb, temp);
  gemm(0, atb, 1024, wo, 1024, nullptr, 1024, ctx, 1024, nullptr, 0, nullptr);

  // z1 / dz / cu input / z_before
  fuse1_kernel<<<16384, 256, 0, stream>>>(z, conn, dzl, corr, ctx, dt, z1, cuin, out_zb);

  // connection update -> out_cn
  gemm(1, cuin, 3072, wc1, 3072, cu_b1, 2048, nullptr, 0,    hbig,    2048, nullptr);
  gemm(0, hbig, 2048, wc2, 2048, cu_b2, 1024, out_cn,  1024, nullptr, 0,    conn);

  // MLP block -> out_z2
  rms_kernel<<<NTOK, 256, 0, stream>>>(z1, w_mlp, z1n, 1024, 0, nullptr, 0, 0);
  gemm(1, z1n,  1024, wm1, 1024, m_b1, 4096, nullptr, 0,    hbig,    4096, nullptr);
  gemm(0, hbig, 4096, wm2, 4096, m_b2, 1024, out_z2,  1024, nullptr, 0,    z1);
}
